// TriangleMultiplicativeUpdate_81655918231689
// MI455X (gfx1250) — compile-verified, matrix operand order checked by reading, not run on hardware
//
#include <hip/hip_runtime.h>
#include <math.h>

typedef __bf16 bf16;
typedef __attribute__((ext_vector_type(16))) __bf16 v16bf;
typedef __attribute__((ext_vector_type(8)))  __bf16 v8bf;
typedef __attribute__((ext_vector_type(8)))  float    v8f;

#define SDIM 512
#define ZDIM 128
#define MDIM (SDIM*SDIM)   // 262144 flattened (s,r) / (s,t)
#define EPSV 1e-5f

__device__ __forceinline__ v16bf cat8(v8bf lo, v8bf hi) {
  return __builtin_shufflevector(lo, hi, 0,1,2,3,4,5,6,7,8,9,10,11,12,13,14,15);
}

// Load a 16x32 bf16 WMMA fragment (A layout; equals B layout of B^T) from LDS.
// ldH = row stride in halves (16B multiple), r0 = base row, k0 = K offset in halves.
// Per CDNA5 ISA 7.12.2: lanes 0-15 row M=lane, elems 0..7 = K0..7, 8..15 = K16..23;
// lanes 16-31 row M=lane-16, elems = K8..15, K24..31.
__device__ __forceinline__ v16bf ldsFrag(const bf16* base, int ldH, int r0, int k0) {
  int lane = threadIdx.x & 31;
  int lo = lane & 15, hi = lane >> 4;
  const bf16* p = base + (size_t)(r0 + lo) * ldH + k0 + hi * 8;
  v8bf a = *(const v8bf*)p;
  v8bf b = *(const v8bf*)(p + 16);
  return cat8(a, b);
}

__device__ __forceinline__ float sigm(float x) { return 1.0f / (1.0f + __expf(-x)); }

// ---------------- Kernel 1: LayerNorm + 5 gated projections -------------------
// Block: 256 threads, 128 rows of x. Outputs:
//   leftT/rightT: bf16 [E][M] (K-contiguous along r for the triangle GEMM)
//   gate:         bf16 [M][Z]
__global__ __launch_bounds__(256) void k_ln_proj(
    const float* __restrict__ pair, const float* __restrict__ lnw, const float* __restrict__ lnb,
    const float* __restrict__ w_l1, const float* __restrict__ w_l2,
    const float* __restrict__ w_r1, const float* __restrict__ w_r2,
    const float* __restrict__ w_g,
    bf16* __restrict__ leftT, bf16* __restrict__ rightT, bf16* __restrict__ gateO)
{
  __shared__ __align__(16) bf16 xs[128 * 136];   // x (LN'd, bf16), padded rows
  __shared__ __align__(16) bf16 wt[64 * 136];    // weight half, transposed [e'][k]
  __shared__ __align__(16) bf16 ts[8][16][16];   // per-wave transpose scratch
  __shared__ float lnwS[128], lnbS[128];

  const int tid  = threadIdx.x;
  const int m0   = blockIdx.x * 128;
  const int wave = tid >> 5, lane = tid & 31;
  const int lo = lane & 15, hi = lane >> 4;

  if (tid < 128) { lnwS[tid] = lnw[tid]; lnbS[tid] = lnb[tid]; }

  // ---- Phase 1: LayerNorm 128 rows (2 threads per row) ----
  {
    int row = tid >> 1, half = tid & 1;
    const float* src = pair + (size_t)(m0 + row) * ZDIM + half * 64;
    float4 vals[16];
    float s = 0.f, ss = 0.f;
#pragma unroll
    for (int i = 0; i < 16; ++i) {
      vals[i] = ((const float4*)src)[i];
      s  += vals[i].x + vals[i].y + vals[i].z + vals[i].w;
      ss += vals[i].x*vals[i].x + vals[i].y*vals[i].y + vals[i].z*vals[i].z + vals[i].w*vals[i].w;
    }
    s  += __shfl_xor(s, 1);
    ss += __shfl_xor(ss, 1);
    float mu  = s * (1.f / 128.f);
    float var = ss * (1.f / 128.f) - mu * mu;
    float rs  = rsqrtf(var + EPSV);
    __syncthreads();  // lnwS ready
    bf16* xrow = xs + row * 136 + half * 64;
#pragma unroll
    for (int i = 0; i < 16; ++i) {
      int c = half * 64 + i * 4;
      xrow[i*4+0] = (bf16)((vals[i].x - mu) * rs * lnwS[c+0] + lnbS[c+0]);
      xrow[i*4+1] = (bf16)((vals[i].y - mu) * rs * lnwS[c+1] + lnbS[c+1]);
      xrow[i*4+2] = (bf16)((vals[i].z - mu) * rs * lnwS[c+2] + lnbS[c+2]);
      xrow[i*4+3] = (bf16)((vals[i].w - mu) * rs * lnwS[c+3] + lnbS[c+3]);
    }
  }
  __syncthreads();

  // Stage 64 output columns of a weight, transposed+bf16: wt[e'][k]
  auto stageW = [&](const float* W, int e0) {
    int k = tid >> 1, h = tid & 1;
    const float* wr = W + (size_t)k * 128 + e0 + h * 32;
#pragma unroll
    for (int j = 0; j < 32; ++j) wt[(h * 32 + j) * 136 + k] = (bf16)wr[j];
  };

  // Wave computes 16(M) x 64(N) via 4 WMMA tiles, K=128
  auto mm = [&](v8f acc[4]) {
#pragma unroll
    for (int kk = 0; kk < 4; ++kk) {
      v16bf a = ldsFrag(xs, 136, wave * 16, kk * 32);
#pragma unroll
      for (int n = 0; n < 4; ++n) {
        v16bf b = ldsFrag(wt, 136, n * 16, kk * 32);
        acc[n] = __builtin_amdgcn_wmma_f32_16x16x32_bf16(
            false, a, false, b, (short)0, acc[n], false, false);
      }
    }
  };

  // ---- left / right gated projections ----
  for (int pr = 0; pr < 2; ++pr) {
    const float* W1 = pr ? w_r1 : w_l1;
    const float* W2 = pr ? w_r2 : w_l2;
    bf16* dst = pr ? rightT : leftT;
    for (int eh = 0; eh < 2; ++eh) {
      __syncthreads();
      stageW(W1, eh * 64);
      __syncthreads();
      v8f acc1[4] = {};
      mm(acc1);
      __syncthreads();
      stageW(W2, eh * 64);
      __syncthreads();
      v8f acc2[4] = {};
      mm(acc2);
#pragma unroll
      for (int n = 0; n < 4; ++n) {
        __syncthreads();
#pragma unroll
        for (int v = 0; v < 8; ++v) {
          float g = sigm(acc1[n][v]);
          ts[wave][lo][v + 8 * hi] = (bf16)(g * acc2[n][v]);  // ts[e][m]
        }
        __syncthreads();
        int er = lane >> 1, hh = lane & 1;
        v8bf outv = *(const v8bf*)&ts[wave][er][hh * 8];
        int eg = eh * 64 + n * 16 + er;
        size_t addr = (size_t)eg * MDIM + (size_t)m0 + wave * 16 + hh * 8;
        *(v8bf*)(dst + addr) = outv;
      }
    }
  }

  // ---- gate projection (natural [M][Z] layout) ----
  for (int eh = 0; eh < 2; ++eh) {
    __syncthreads();
    stageW(w_g, eh * 64);
    __syncthreads();
    v8f acc[4] = {};
    mm(acc);
#pragma unroll
    for (int n = 0; n < 4; ++n) {
#pragma unroll
      for (int v = 0; v < 8; ++v) {
        int mrow = m0 + wave * 16 + v + 8 * hi;
        int ecol = eh * 64 + n * 16 + lo;
        gateO[(size_t)mrow * ZDIM + ecol] = (bf16)sigm(acc[n][v]);
      }
    }
  }
}

// ---------------- Kernel 2: triangle einsum (batched GEMM over e) -------------
// grid = (t_tile=8, s_tile=4, e=128); block = 256 (8 waves), C tile 128x64.
// Each wave owns a 2x2 grid of 16x16 tiles: 4 fragment loads -> 4 WMMAs per K-step.
__global__ __launch_bounds__(256) void k_tri_gemm(
    const bf16* __restrict__ leftT, const bf16* __restrict__ rightT, float* __restrict__ pOut)
{
  __shared__ __align__(16) bf16 As[128 * 40];  // 128 rows x 32 K halves, padded to 80B
  __shared__ __align__(16) bf16 Bs[64 * 40];   //  64 rows x 32 K halves

  const int tid = threadIdx.x;
  const int tb = blockIdx.x, sb = blockIdx.y, e = blockIdx.z;
  const bf16* Abase = leftT  + (size_t)e * MDIM + (size_t)(sb * 128) * SDIM;
  const bf16* Bbase = rightT + (size_t)e * MDIM + (size_t)(tb * 64)  * SDIM;

  const int wave = tid >> 5, lane = tid & 31;
  const int s0 = (wave & 3) * 2;   // wave's 2 s subtiles (of 8)
  const int t0 = (wave >> 2) * 2;  // wave's 2 t subtiles (of 4)
  v8f acc[2][2] = {};

  // cooperative-load mappings
  const int ar = tid >> 1, ac = (tid & 1) * 2;  // As: 128 rows x 2 chunks/thread
  const int br = tid >> 2, bc = tid & 3;        // Bs: 64 rows x 1 chunk/thread

  for (int k0 = 0; k0 < SDIM; k0 += 32) {
    __syncthreads();
    v8bf a0 = *(const v8bf*)(Abase + (size_t)ar * SDIM + k0 + ac * 8);
    v8bf a1 = *(const v8bf*)(Abase + (size_t)ar * SDIM + k0 + ac * 8 + 8);
    v8bf bv = *(const v8bf*)(Bbase + (size_t)br * SDIM + k0 + bc * 8);
    *(v8bf*)&As[ar * 40 + ac * 8]     = a0;
    *(v8bf*)&As[ar * 40 + ac * 8 + 8] = a1;
    *(v8bf*)&Bs[br * 40 + bc * 8]     = bv;
    __syncthreads();
    v16bf af[2], bf[2];
#pragma unroll
    for (int i = 0; i < 2; ++i) af[i] = ldsFrag(As, 40, (s0 + i) * 16, 0);
#pragma unroll
    for (int j = 0; j < 2; ++j) bf[j] = ldsFrag(Bs, 40, (t0 + j) * 16, 0);
#pragma unroll
    for (int i = 0; i < 2; ++i)
#pragma unroll
      for (int j = 0; j < 2; ++j)
        acc[i][j] = __builtin_amdgcn_wmma_f32_16x16x32_bf16(
            false, af[i], false, bf[j], (short)0, acc[i][j], false, false);
  }

  const int lo = lane & 15, hi = lane >> 4;
#pragma unroll
  for (int i = 0; i < 2; ++i) {
#pragma unroll
    for (int j = 0; j < 2; ++j) {
#pragma unroll
      for (int v = 0; v < 8; ++v) {
        int sg   = sb * 128 + (s0 + i) * 16 + v + 8 * hi;
        int tcol = tb * 64  + (t0 + j) * 16 + lo;   // lanes 0-15 contiguous
        pOut[(size_t)e * MDIM + (size_t)sg * SDIM + tcol] = acc[i][j][v];
      }
    }
  }
}

// ---------------- Kernel 3: LN over E + out_w + gate + residual ---------------
// Block handles 64 rows of flattened (s,t); p stored [E][M] so stats reads coalesce.
__global__ __launch_bounds__(256) void k_out(
    const float* __restrict__ pIn,      // [E][M] fp32
    const float* __restrict__ normw, const float* __restrict__ normb,
    const float* __restrict__ outw,     // [E][Z] fp32
    const bf16*  __restrict__ gate,     // [M][Z] bf16
    const float* __restrict__ residual, // [M][Z] fp32
    float* __restrict__ out)
{
  __shared__ __align__(16) bf16 pbf[64 * 136];   // LN'd p, bf16 [m][e]
  __shared__ __align__(16) bf16 wt[128 * 136];   // out_w^T bf16 [z][e]
  __shared__ float red[64][4][2];
  __shared__ float muS[64], rsS[64];
  __shared__ float nwS[128], nbS[128];

  const int tid = threadIdx.x;
  const size_t m0 = (size_t)blockIdx.x * 64;
  if (tid < 128) { nwS[tid] = normw[tid]; nbS[tid] = normb[tid]; }

  const int col = tid & 63, q = tid >> 6;   // 4 threads per column, e quarters
  float s = 0.f, ss = 0.f;
  for (int e = q * 32; e < q * 32 + 32; ++e) {
    float v = pIn[(size_t)e * MDIM + m0 + col];
    s += v; ss += v * v;
  }
  red[col][q][0] = s; red[col][q][1] = ss;
  __syncthreads();
  if (tid < 64) {
    float S_ = 0.f, SS = 0.f;
    for (int q2 = 0; q2 < 4; ++q2) { S_ += red[tid][q2][0]; SS += red[tid][q2][1]; }
    float mu = S_ * (1.f / 128.f);
    float var = SS * (1.f / 128.f) - mu * mu;
    muS[tid] = mu; rsS[tid] = rsqrtf(var + EPSV);
  }
  __syncthreads();
  {
    float mu = muS[col], rs = rsS[col];
    for (int e = q * 32; e < q * 32 + 32; ++e) {
      float v = pIn[(size_t)e * MDIM + m0 + col];
      pbf[col * 136 + e] = (bf16)((v - mu) * rs * nwS[e] + nbS[e]);
    }
  }
  {
    int k = tid >> 1, h = tid & 1;          // stage out_w transposed
    const float* wr = outw + (size_t)k * ZDIM + h * 64;
#pragma unroll 8
    for (int j = 0; j < 64; ++j) wt[(h * 64 + j) * 136 + k] = (bf16)wr[j];
  }
  __syncthreads();

  const int wave = tid >> 5, lane = tid & 31, lo = lane & 15, hi = lane >> 4;
  const int msub = wave & 3, nb0 = (wave >> 2) * 4;
  v8f acc[4] = {};
#pragma unroll
  for (int kk = 0; kk < 4; ++kk) {
    v16bf a = ldsFrag(pbf, 136, msub * 16, kk * 32);
#pragma unroll
    for (int n = 0; n < 4; ++n) {
      v16bf b = ldsFrag(wt, 136, (nb0 + n) * 16, kk * 32);
      acc[n] = __builtin_amdgcn_wmma_f32_16x16x32_bf16(
          false, a, false, b, (short)0, acc[n], false, false);
    }
  }
#pragma unroll
  for (int n = 0; n < 4; ++n) {
#pragma unroll
    for (int v = 0; v < 8; ++v) {
      size_t mrow = m0 + msub * 16 + v + 8 * hi;
      int z = (nb0 + n) * 16 + lo;
      size_t addr = mrow * ZDIM + z;
      float g = (float)gate[addr];
      out[addr] = residual[addr] + acc[n][v] * g;
    }
  }
}

extern "C" void kernel_launch(void* const* d_in, const int* in_sizes, int n_in,
                              void* d_out, int out_size, void* d_ws, size_t ws_size,
                              hipStream_t stream) {
  const float* pair = (const float*)d_in[0];
  const float* lnw  = (const float*)d_in[1];
  const float* lnb  = (const float*)d_in[2];
  const float* l1   = (const float*)d_in[3];
  const float* l2   = (const float*)d_in[4];
  const float* r1   = (const float*)d_in[5];
  const float* r2   = (const float*)d_in[6];
  const float* nw   = (const float*)d_in[7];
  const float* nb   = (const float*)d_in[8];
  const float* ow   = (const float*)d_in[9];
  const float* gw   = (const float*)d_in[10];

  char* ws = (char*)d_ws;
  bf16*  leftT  = (bf16*)(ws);                         // 64 MiB
  bf16*  rightT = (bf16*)(ws + ((size_t)64  << 20));   // 64 MiB
  bf16*  gateB  = (bf16*)(ws + ((size_t)128 << 20));   // 64 MiB
  float* pBuf   = (float*)(ws + ((size_t)192 << 20));  // 128 MiB

  k_ln_proj<<<MDIM / 128, 256, 0, stream>>>(pair, lnw, lnb, l1, l2, r1, r2, gw,
                                            leftT, rightT, gateB);
  k_tri_gemm<<<dim3(8, 4, 128), 256, 0, stream>>>(leftT, rightT, pBuf);
  k_out<<<MDIM / 64, 256, 0, stream>>>(pBuf, nw, nb, ow, gateB, pair, (float*)d_out);
}
